// QBitsLinear_33638183862750
// MI455X (gfx1250) — matrix operand order repaired and checked by reading, not compiled, not run on hardware
//
#include <hip/hip_runtime.h>
#include <hip/hip_bf16.h>

typedef __attribute__((ext_vector_type(16))) _Float16 v16h;
typedef __attribute__((ext_vector_type(8)))  _Float16 v8h;
typedef __attribute__((ext_vector_type(8)))  float    v8f;

#define IN_FEATS   4096
#define OUT_FEATS  14336
#define GROUP_SIZE 128
#define N_TOKENS   64
#define KSPLIT     4                         // K split across blockIdx.y
#define KCHUNK     (IN_FEATS / KSPLIT)       // 1024
#define GROUPS_PER_CHUNK (KCHUNK / GROUP_SIZE) // 8

// ---------------------------------------------------------------------------
// Kernel 1: convert x (fp32) -> fp16 packed row-major, and per-token row sums
// One block per token, 256 threads, 16 elements each (256*16 = 4096).
// ---------------------------------------------------------------------------
__global__ __launch_bounds__(256) void prep_x_kernel(const float* __restrict__ x,
                                                     _Float16* __restrict__ x16,
                                                     float* __restrict__ xsum) {
    __shared__ float red[256];
    const int t   = blockIdx.x;            // token
    const int tid = threadIdx.x;
    const float* xr = x + (size_t)t * IN_FEATS + tid * 16;
    _Float16*   xo = x16 + (size_t)t * IN_FEATS + tid * 16;
    float s = 0.f;
#pragma unroll
    for (int i = 0; i < 16; ++i) {
        float v = xr[i];
        s += v;
        xo[i] = (_Float16)v;
    }
    red[tid] = s;
    __syncthreads();
#pragma unroll
    for (int off = 128; off > 0; off >>= 1) {
        if (tid < off) red[tid] += red[tid + off];
        __syncthreads();
    }
    if (tid == 0) xsum[t] = red[0];
}

// ---------------------------------------------------------------------------
// Kernel 2: out[t,o] = bias[o] + xsum[t] * u[o]   (also un-poisons d_out)
// ---------------------------------------------------------------------------
__global__ __launch_bounds__(256) void init_out_kernel(const float* __restrict__ bias,
                                                       const float* __restrict__ u,
                                                       const float* __restrict__ xsum,
                                                       float* __restrict__ out) {
    int idx = blockIdx.x * 256 + threadIdx.x;
    if (idx >= N_TOKENS * OUT_FEATS) return;
    int t = idx / OUT_FEATS;
    int o = idx - t * OUT_FEATS;
    out[idx] = bias[o] + xsum[t] * u[o];
}

// ---------------------------------------------------------------------------
// Kernel 3: main quantized GEMM with WMMA f16 -> f32.
//   block = 128 threads = 4 wave32s; wave w handles columns
//   [blockIdx.x*64 + w*16, +16). blockIdx.y selects a 1024-wide K chunk.
//   Per 128-K group: 4x V_WMMA_F32_16X16X32_F16 into a group accumulator
//   (weights are exact small ints in f16), then one scale-FMA per element.
//   Partial results combined across K-chunks with global_atomic_add_f32.
// ---------------------------------------------------------------------------
__global__ __launch_bounds__(128) void qgemm_wmma_kernel(const int*   __restrict__ W,
                                                         const float* __restrict__ scales,
                                                         const _Float16* __restrict__ x16,
                                                         float* __restrict__ out) {
    const int wave = threadIdx.x >> 5;
    const int lane = threadIdx.x & 31;
    const int nlo  = lane & 15;          // N column within tile / M row for A
    const int hi   = (lane >> 4) & 1;    // half-wave selector (K split)

    const int n0   = blockIdx.x * 64 + wave * 16;
    const int col  = n0 + nlo;
    const int k0   = blockIdx.y * KCHUNK;

    v8f acc[4];
#pragma unroll
    for (int m = 0; m < 4; ++m) acc[m] = (v8f)0.f;

    for (int g = 0; g < GROUPS_PER_CHUNK; ++g) {
        const int gk = k0 + g * GROUP_SIZE;
        const float s = scales[(size_t)(gk / GROUP_SIZE) * OUT_FEATS + col];

        // prefetch next group's weight rows for this lane's column
        if (g + 1 < GROUPS_PER_CHUNK) {
            const int* np = W + (size_t)(gk + GROUP_SIZE + hi * 16) * OUT_FEATS + col;
            __builtin_prefetch(np, 0, 1);
        }

        v8f accg[4];
#pragma unroll
        for (int m = 0; m < 4; ++m) accg[m] = (v8f)0.f;

#pragma unroll
        for (int ks = 0; ks < GROUP_SIZE / 32; ++ks) {
            const int kb = gk + ks * 32;

            // ---- B fragment: 32x16 f16. Lanes 0-15 hold K=0..15, lanes
            // 16-31 hold K=16..31; half h of the 8 VGPRs is K = h (+16 for hi).
            // Raw int4 values converted to f16 exactly (no dequant here).
            const int* wp = W + (size_t)(kb + hi * 16) * OUT_FEATS + col;
            v16h b;
#pragma unroll
            for (int h = 0; h < 16; ++h)
                b[h] = (_Float16)(float)wp[(size_t)h * OUT_FEATS];

            // ---- A fragments + WMMA per 16-row M tile
#pragma unroll
            for (int m = 0; m < 4; ++m) {
                const _Float16* xp =
                    x16 + (size_t)(m * 16 + nlo) * IN_FEATS + kb + hi * 16;
                v8h alo = *(const v8h*)xp;          // K 0..7   (or 16..23)
                v8h ahi = *(const v8h*)(xp + 8);    // K 8..15  (or 24..31)
                v16h a;
#pragma unroll
                for (int h = 0; h < 8; ++h) { a[h] = alo[h]; a[h + 8] = ahi[h]; }

                accg[m] = __builtin_amdgcn_wmma_f32_16x16x32_f16(
                    /*neg_a=*/false, a, /*neg_b=*/false, b,
                    /*c_mod=*/(short)0, accg[m],
                    /*reuse_a=*/false, /*reuse_b=*/false);
            }
        }

        // fold the group accumulator in with its per-(group,col) scale
#pragma unroll
        for (int m = 0; m < 4; ++m)
#pragma unroll
            for (int r = 0; r < 8; ++r)
                acc[m][r] += s * accg[m][r];
    }

    // ---- writeback: C/D layout -> row = m*16 + hi*8 + r, col = n0 + nlo
#pragma unroll
    for (int m = 0; m < 4; ++m)
#pragma unroll
        for (int r = 0; r < 8; ++r) {
            int row = m * 16 + hi * 8 + r;
            atomicAdd(out + (size_t)row * OUT_FEATS + col, acc[m][r]);
        }
}

// ---------------------------------------------------------------------------
// Launch
// ---------------------------------------------------------------------------
extern "C" void kernel_launch(void* const* d_in, const int* in_sizes, int n_in,
                              void* d_out, int out_size, void* d_ws, size_t ws_size,
                              hipStream_t stream) {
    const float* x      = (const float*)d_in[0];
    const int*   W_int  = (const int*)d_in[1];
    const float* scales = (const float*)d_in[2];
    const float* u      = (const float*)d_in[3];
    const float* bias   = (const float*)d_in[4];
    float*       out    = (float*)d_out;

    // workspace: [x16: 64*4096 f16 = 512 KB][xsum: 64 f32]
    _Float16* x16  = (_Float16*)d_ws;
    float*    xsum = (float*)((char*)d_ws + (size_t)N_TOKENS * IN_FEATS * sizeof(_Float16));

    prep_x_kernel<<<N_TOKENS, 256, 0, stream>>>(x, x16, xsum);

    const int total = N_TOKENS * OUT_FEATS;
    init_out_kernel<<<(total + 255) / 256, 256, 0, stream>>>(bias, u, xsum, out);

    dim3 grid(OUT_FEATS / 64, KSPLIT);   // 224 x 4 blocks, 4 waves each
    qgemm_wmma_kernel<<<grid, 128, 0, stream>>>(W_int, scales, x16, out);
}
